// KoLeoLoss_Contrastive_9955734192550
// MI455X (gfx1250) — matrix operand order repaired and checked by reading, not compiled, not run on hardware
//
#include <hip/hip_runtime.h>
#include <math.h>

// Problem constants (reference: anchor[8192,1024], sample[8192,1024], f32)
#define KN   8192      // anchors (= rows), also sample count
#define KD   1024      // feature dim
#define KTOT 16384     // total columns = anchors ++ samples

typedef __attribute__((ext_vector_type(2)))  float  v2f;    // f32 WMMA A/B fragment
typedef __attribute__((ext_vector_type(8)))  float  v8f;    // WMMA C/D fragment (16x16 f32)
typedef __attribute__((ext_vector_type(8)))  __bf16 v8bf;   // one b128 of bf16
typedef __attribute__((ext_vector_type(16))) __bf16 v16bf;  // bf16 WMMA A/B fragment (16x32 / 32x16)

// ---------------------------------------------------------------------------
// Kernel 1: exact f32 row norms ||x||^2, anchor rows then sample rows.
// ---------------------------------------------------------------------------
__global__ __launch_bounds__(256) void koleo_norms(const float* __restrict__ A,
                                                   const float* __restrict__ S,
                                                   float* __restrict__ norm) {
  const int wave = threadIdx.x >> 5;
  const int lane = threadIdx.x & 31;
  const int row  = blockIdx.x * 8 + wave;          // grid = KTOT/8
  const float* src = (row < KN) ? (A + (size_t)row * KD)
                                : (S + (size_t)(row - KN) * KD);
  float s = 0.0f;
  for (int k = lane * 4; k < KD; k += 128) {
    const float4 x = *(const float4*)(src + k);
    s = fmaf(x.x, x.x, s);
    s = fmaf(x.y, x.y, s);
    s = fmaf(x.z, x.z, s);
    s = fmaf(x.w, x.w, s);
  }
  #pragma unroll
  for (int off = 16; off; off >>= 1) s += __shfl_xor(s, off, 32);
  if (lane == 0) norm[row] = s;
}

// ---------------------------------------------------------------------------
// Kernel 2: seed per-row min buffer with +inf bits (nonneg float order == uint
// order, so atomicMin on bits implements float min).
// ---------------------------------------------------------------------------
__global__ void koleo_init(unsigned int* __restrict__ minbits) {
  minbits[blockIdx.x * blockDim.x + threadIdx.x] = 0x7F800000u;  // +inf
}

// ---------------------------------------------------------------------------
// Kernel 3: split f32 -> bf16 hi/lo (bf16x3 decomposition).
//   hi = top 16 bits of f32 (truncation) -> lo = x - hi is EXACT in f32,
//   then lo is rounded (RNE) to bf16. x == hi + lo up to 2^-16 rel.
// Each thread converts 4 contiguous floats; packed dword stores.
// ---------------------------------------------------------------------------
__global__ __launch_bounds__(256) void koleo_split(const float* __restrict__ X,
                                                   unsigned int* __restrict__ hi,
                                                   unsigned int* __restrict__ lo) {
  const size_t q = (size_t)blockIdx.x * 256 + threadIdx.x;   // quad index
  const float4 x = ((const float4*)X)[q];
  const unsigned bx0 = __float_as_uint(x.x), bx1 = __float_as_uint(x.y);
  const unsigned bx2 = __float_as_uint(x.z), bx3 = __float_as_uint(x.w);
  const unsigned h01 = (bx0 >> 16) | (bx1 & 0xffff0000u);
  const unsigned h23 = (bx2 >> 16) | (bx3 & 0xffff0000u);
  const float l0 = x.x - __uint_as_float(bx0 & 0xffff0000u);
  const float l1 = x.y - __uint_as_float(bx1 & 0xffff0000u);
  const float l2 = x.z - __uint_as_float(bx2 & 0xffff0000u);
  const float l3 = x.w - __uint_as_float(bx3 & 0xffff0000u);
  const unsigned s0 = __builtin_bit_cast(unsigned short, (__bf16)l0);
  const unsigned s1 = __builtin_bit_cast(unsigned short, (__bf16)l1);
  const unsigned s2 = __builtin_bit_cast(unsigned short, (__bf16)l2);
  const unsigned s3 = __builtin_bit_cast(unsigned short, (__bf16)l3);
  uint2 hv; hv.x = h01;               hv.y = h23;
  uint2 lv; lv.x = s0 | (s1 << 16);   lv.y = s2 | (s3 << 16);
  ((uint2*)hi)[q] = hv;
  ((uint2*)lo)[q] = lv;
}

// ---------------------------------------------------------------------------
// bf16 fragment loaders (wave32; lh = lane>>4 offset is pre-added to rowp).
// A 16x32 (ISA table): lane holds K = {k..k+7, k+16..k+23} (+8 for lanes>=16).
// B 32x16: same K order as A: lane holds K = {k..k+7, k+16..k+23} (+8 for lanes>=16).
// ---------------------------------------------------------------------------
__device__ __forceinline__ v16bf load_a_frag(const __bf16* rowp, int k) {
  const v8bf p0 = *(const v8bf*)(rowp + k);
  const v8bf p1 = *(const v8bf*)(rowp + k + 16);
  v16bf f;
  #pragma unroll
  for (int i = 0; i < 8; ++i) { f[i] = p0[i]; f[i + 8] = p1[i]; }
  return f;
}
__device__ __forceinline__ v16bf load_b_frag(const __bf16* rowp, int k) {
  const v8bf p0 = *(const v8bf*)(rowp + k);
  const v8bf p1 = *(const v8bf*)(rowp + k + 16);
  v16bf f;
  #pragma unroll
  for (int i = 0; i < 8; ++i) { f[i] = p0[i]; f[i + 8] = p1[i]; }
  return f;
}

#define WMMA_BF16(A_, B_, C_) \
  __builtin_amdgcn_wmma_f32_16x16x32_bf16(false, (A_), false, (B_), (short)0, (C_), false, false)

// ---------------------------------------------------------------------------
// Kernel 4 (main): bf16x3 WMMA GEMM + fused row-min of squared distances.
//   dot(x,y) ~= xh.yh + xh.yl + xl.yh  -> 3x v_wmma_f32_16x16x32_bf16 per K=32
//   (vs 8x v_wmma_f32_16x16x4_f32: ~2.7x less matrix-pipe work, same bytes).
// Tiling: 256 thr = 8 waves; wave -> 16x64 tile (4 accumulators, A reused 4x);
// block -> 64 rows x 128 cols; grid = (KN/64 row blocks, 8 col slices of 2048).
// C/D layout: acc[v] is (row = rowBase + v + 8*(lane>>4), col = colBase+lane&15).
// ---------------------------------------------------------------------------
__global__ __launch_bounds__(256) void koleo_gemm_min_bf16(
    const __bf16* __restrict__ Ahi, const __bf16* __restrict__ Alo,
    const __bf16* __restrict__ Shi, const __bf16* __restrict__ Slo,
    const float* __restrict__ norm, unsigned int* __restrict__ minbits) {
  const int tid  = threadIdx.x;
  const int wave = tid >> 5;
  const int lane = tid & 31;
  const int l15  = lane & 15;
  const int lh   = lane >> 4;
  const int rowTile = wave >> 1;       // 0..3
  const int colHalf = wave & 1;        // 0..1

  const int rowBase   = blockIdx.x * 64 + rowTile * 16;
  const int sliceBase = blockIdx.y * 2048;

  const __bf16* aHiRow = Ahi + (size_t)(rowBase + l15) * KD + lh * 8;
  const __bf16* aLoRow = Alo + (size_t)(rowBase + l15) * KD + lh * 8;

  float naRow[8];
  #pragma unroll
  for (int v = 0; v < 8; ++v) naRow[v] = norm[rowBase + v + 8 * lh];

  float rmin[8];
  #pragma unroll
  for (int v = 0; v < 8; ++v) rmin[v] = 3.402823466e38f;

  for (int ct = 0; ct < 2048; ct += 128) {
    const int colBase = sliceBase + ct + colHalf * 64;   // 64-col group never straddles KN
    const __bf16* Yhi; const __bf16* Ylo; int cb;
    if (colBase < KN) { Yhi = Ahi; Ylo = Alo; cb = colBase; }
    else              { Yhi = Shi; Ylo = Slo; cb = colBase - KN; }

    const __bf16* bHi0 = Yhi + (size_t)(cb + l15) * KD + lh * 8;
    const __bf16* bLo0 = Ylo + (size_t)(cb + l15) * KD + lh * 8;
    const size_t tstep = (size_t)16 * KD;

    v8f acc0 = {0.f,0.f,0.f,0.f,0.f,0.f,0.f,0.f};
    v8f acc1 = acc0, acc2 = acc0, acc3 = acc0;

    for (int k = 0; k < KD; k += 32) {
      const v16bf ah  = load_a_frag(aHiRow, k);
      const v16bf al  = load_a_frag(aLoRow, k);
      const v16bf bh0 = load_b_frag(bHi0 + 0 * tstep, k);
      const v16bf bh1 = load_b_frag(bHi0 + 1 * tstep, k);
      const v16bf bh2 = load_b_frag(bHi0 + 2 * tstep, k);
      const v16bf bh3 = load_b_frag(bHi0 + 3 * tstep, k);
      const v16bf bl0 = load_b_frag(bLo0 + 0 * tstep, k);
      const v16bf bl1 = load_b_frag(bLo0 + 1 * tstep, k);
      const v16bf bl2 = load_b_frag(bLo0 + 2 * tstep, k);
      const v16bf bl3 = load_b_frag(bLo0 + 3 * tstep, k);
      acc0 = WMMA_BF16(ah, bh0, acc0);
      acc1 = WMMA_BF16(ah, bh1, acc1);
      acc2 = WMMA_BF16(ah, bh2, acc2);
      acc3 = WMMA_BF16(ah, bh3, acc3);
      acc0 = WMMA_BF16(ah, bl0, acc0);
      acc1 = WMMA_BF16(ah, bl1, acc1);
      acc2 = WMMA_BF16(ah, bl2, acc2);
      acc3 = WMMA_BF16(ah, bl3, acc3);
      acc0 = WMMA_BF16(al, bh0, acc0);
      acc1 = WMMA_BF16(al, bh1, acc1);
      acc2 = WMMA_BF16(al, bh2, acc2);
      acc3 = WMMA_BF16(al, bh3, acc3);
    }

    // Epilogue: sq = ||x||^2 + ||y||^2 - 2 x.y ; exclude self pair (j==i, only
    // possible in the anchor half since global cols >= KN never equal i < KN)
    // and any sq<=0 (mirrors the reference's d==0 -> excluded-from-min rule).
    const float nc0 = norm[colBase +  0 + l15];
    const float nc1 = norm[colBase + 16 + l15];
    const float nc2 = norm[colBase + 32 + l15];
    const float nc3 = norm[colBase + 48 + l15];
    const int   j0  = colBase +  0 + l15;
    const int   j1  = colBase + 16 + l15;
    const int   j2  = colBase + 32 + l15;
    const int   j3  = colBase + 48 + l15;

    #pragma unroll
    for (int v = 0; v < 8; ++v) {
      const int   i  = rowBase + v + 8 * lh;
      const float na = naRow[v];
      float sq;
      sq = na + nc0 - 2.0f * acc0[v]; if (sq > 0.0f && j0 != i) rmin[v] = fminf(rmin[v], sq);
      sq = na + nc1 - 2.0f * acc1[v]; if (sq > 0.0f && j1 != i) rmin[v] = fminf(rmin[v], sq);
      sq = na + nc2 - 2.0f * acc2[v]; if (sq > 0.0f && j2 != i) rmin[v] = fminf(rmin[v], sq);
      sq = na + nc3 - 2.0f * acc3[v]; if (sq > 0.0f && j3 != i) rmin[v] = fminf(rmin[v], sq);
    }
  }

  #pragma unroll
  for (int v = 0; v < 8; ++v) {
    float m = rmin[v];
    #pragma unroll
    for (int off = 8; off; off >>= 1) m = fminf(m, __shfl_xor(m, off, 32));
    if (l15 == 0)
      atomicMin(&minbits[rowBase + v + 8 * lh], __float_as_uint(m));
  }
}

// ---------------------------------------------------------------------------
// Fallback: exact-f32 WMMA GEMM (V_WMMA_F32_16X16X4_F32), used only if d_ws is
// too small for the bf16 hi/lo split buffers. Same tiling and epilogue.
// ---------------------------------------------------------------------------
__global__ __launch_bounds__(256) void koleo_gemm_min_f32(const float* __restrict__ A,
                                                          const float* __restrict__ S,
                                                          const float* __restrict__ norm,
                                                          unsigned int* __restrict__ minbits) {
  const int tid  = threadIdx.x;
  const int wave = tid >> 5;
  const int lane = tid & 31;
  const int l15  = lane & 15;
  const int lh   = lane >> 4;
  const int rowTile = wave >> 1;
  const int colHalf = wave & 1;

  const int rowBase   = blockIdx.x * 64 + rowTile * 16;
  const int sliceBase = blockIdx.y * 2048;

  const float* aRow = A + (size_t)(rowBase + l15) * KD + lh * 2;

  float naRow[8];
  #pragma unroll
  for (int v = 0; v < 8; ++v) naRow[v] = norm[rowBase + v + 8 * lh];

  float rmin[8];
  #pragma unroll
  for (int v = 0; v < 8; ++v) rmin[v] = 3.402823466e38f;

  for (int ct = 0; ct < 2048; ct += 128) {
    const int colBase = sliceBase + ct + colHalf * 64;
    const float* Y; int cb;
    if (colBase < KN) { Y = A; cb = colBase; }
    else              { Y = S; cb = colBase - KN; }

    const float* bRow0 = Y + (size_t)(cb + l15) * KD + lh * 2;
    const float* bRow1 = bRow0 + (size_t)16 * KD;
    const float* bRow2 = bRow0 + (size_t)32 * KD;
    const float* bRow3 = bRow0 + (size_t)48 * KD;

    v8f acc0 = {0.f,0.f,0.f,0.f,0.f,0.f,0.f,0.f};
    v8f acc1 = acc0, acc2 = acc0, acc3 = acc0;

    #pragma unroll 4
    for (int k = 0; k < KD; k += 4) {
      const v2f a  = *(const v2f*)(aRow  + k);
      const v2f b0 = *(const v2f*)(bRow0 + k);
      const v2f b1 = *(const v2f*)(bRow1 + k);
      const v2f b2 = *(const v2f*)(bRow2 + k);
      const v2f b3 = *(const v2f*)(bRow3 + k);
      acc0 = __builtin_amdgcn_wmma_f32_16x16x4_f32(false, a, false, b0, (short)0, acc0, false, false);
      acc1 = __builtin_amdgcn_wmma_f32_16x16x4_f32(false, a, false, b1, (short)0, acc1, false, false);
      acc2 = __builtin_amdgcn_wmma_f32_16x16x4_f32(false, a, false, b2, (short)0, acc2, false, false);
      acc3 = __builtin_amdgcn_wmma_f32_16x16x4_f32(false, a, false, b3, (short)0, acc3, false, false);
    }

    const float nc0 = norm[colBase +  0 + l15];
    const float nc1 = norm[colBase + 16 + l15];
    const float nc2 = norm[colBase + 32 + l15];
    const float nc3 = norm[colBase + 48 + l15];
    const int   j0  = colBase +  0 + l15;
    const int   j1  = colBase + 16 + l15;
    const int   j2  = colBase + 32 + l15;
    const int   j3  = colBase + 48 + l15;

    #pragma unroll
    for (int v = 0; v < 8; ++v) {
      const int   i  = rowBase + v + 8 * lh;
      const float na = naRow[v];
      float sq;
      sq = na + nc0 - 2.0f * acc0[v]; if (sq > 0.0f && j0 != i) rmin[v] = fminf(rmin[v], sq);
      sq = na + nc1 - 2.0f * acc1[v]; if (sq > 0.0f && j1 != i) rmin[v] = fminf(rmin[v], sq);
      sq = na + nc2 - 2.0f * acc2[v]; if (sq > 0.0f && j2 != i) rmin[v] = fminf(rmin[v], sq);
      sq = na + nc3 - 2.0f * acc3[v]; if (sq > 0.0f && j3 != i) rmin[v] = fminf(rmin[v], sq);
    }
  }

  #pragma unroll
  for (int v = 0; v < 8; ++v) {
    float m = rmin[v];
    #pragma unroll
    for (int off = 8; off; off >>= 1) m = fminf(m, __shfl_xor(m, off, 32));
    if (l15 == 0)
      atomicMin(&minbits[rowBase + v + 8 * lh], __float_as_uint(m));
  }
}

// ---------------------------------------------------------------------------
// Kernel 5: loss = -(1/N) * sum_i log(sqrt(min_sq_i)) = -(1/N) * sum 0.5*log(sq)
// ---------------------------------------------------------------------------
__global__ __launch_bounds__(256) void koleo_loss(const unsigned int* __restrict__ minbits,
                                                  float* __restrict__ out) {
  __shared__ float red[256];
  float s = 0.0f;
  for (int i = threadIdx.x; i < KN; i += 256) {
    const float sq = __uint_as_float(minbits[i]);
    s += 0.5f * logf(sq);
  }
  red[threadIdx.x] = s;
  __syncthreads();
  for (int off = 128; off; off >>= 1) {
    if (threadIdx.x < off) red[threadIdx.x] += red[threadIdx.x + off];
    __syncthreads();
  }
  if (threadIdx.x == 0) out[0] = -red[0] / (float)KN;
}

// ---------------------------------------------------------------------------
// Launcher. Workspace layout (bytes):
//   [0,        65536)   norm[16384] (f32)
//   [65536,    98304)   minbits[8192] (u32)
//   [98304,  +16MB)     Ahi  bf16[8192][1024]
//   ...                 Alo, Shi, Slo (16MB each)  -> total 98304 + 64MB
// Falls back to exact-f32 WMMA path if ws_size is insufficient.
// ---------------------------------------------------------------------------
extern "C" void kernel_launch(void* const* d_in, const int* in_sizes, int n_in,
                              void* d_out, int out_size, void* d_ws, size_t ws_size,
                              hipStream_t stream) {
  (void)in_sizes; (void)n_in; (void)out_size;
  const float* anchor = (const float*)d_in[0];
  const float* sample = (const float*)d_in[1];
  float* out = (float*)d_out;

  char* ws = (char*)d_ws;
  float*        norm    = (float*)ws;
  unsigned int* minbits = (unsigned int*)(ws + (size_t)KTOT * 4);
  const size_t  matB    = (size_t)KN * KD * 2;          // one bf16 matrix
  char*         splitB  = ws + 98304;
  unsigned int* AhiW = (unsigned int*)(splitB + 0 * matB);
  unsigned int* AloW = (unsigned int*)(splitB + 1 * matB);
  unsigned int* ShiW = (unsigned int*)(splitB + 2 * matB);
  unsigned int* SloW = (unsigned int*)(splitB + 3 * matB);
  const size_t need = 98304 + 4 * matB;

  koleo_norms<<<KTOT / 8, 256, 0, stream>>>(anchor, sample, norm);
  koleo_init<<<KN / 256, 256, 0, stream>>>(minbits);

  if (ws_size >= need) {
    const int splitBlocks = (KN * KD / 4) / 256;        // 4 floats per thread
    koleo_split<<<splitBlocks, 256, 0, stream>>>(anchor, AhiW, AloW);
    koleo_split<<<splitBlocks, 256, 0, stream>>>(sample, ShiW, SloW);
    koleo_gemm_min_bf16<<<dim3(KN / 64, 8), 256, 0, stream>>>(
        (const __bf16*)AhiW, (const __bf16*)AloW,
        (const __bf16*)ShiW, (const __bf16*)SloW, norm, minbits);
  } else {
    koleo_gemm_min_f32<<<dim3(KN / 64, 8), 256, 0, stream>>>(anchor, sample, norm, minbits);
  }

  koleo_loss<<<1, 256, 0, stream>>>(minbits, out);
}
